// GroupQueryAttention_51565377356068
// MI455X (gfx1250) — compile-verified, matrix operand order checked by reading, not run on hardware
//
#include <hip/hip_runtime.h>
#include <hip/hip_bf16.h>

// ---------------------------------------------------------------------------
// GQA forward for MI455X (gfx1250, wave32, WMMA, TDM).
//   fp32 -> bf16 convert, TDM/LDS-staged WMMA GEMM projections,
//   flash-style WMMA attention (fp32 online softmax),
//   WMMA output projection (+bias, fp32).
// ---------------------------------------------------------------------------

#define S_LEN    2048
#define MDIM     2048
#define NHEADS   32
#define HDIM     64
#define NKV      8
#define KVDIM    (NKV * HDIM)   // 512

#define GEMM_KC   32            // K per staged tile
#define GEMM_PAD  40            // LDS row pitch (elements): 32 data + 8 pad

typedef __attribute__((ext_vector_type(16))) __bf16 v16bf;
typedef __attribute__((ext_vector_type(8)))  float  v8f;
typedef unsigned int u32x4 __attribute__((ext_vector_type(4)));
typedef int          i32x8 __attribute__((ext_vector_type(8)));
typedef int          i32x4 __attribute__((ext_vector_type(4)));

#if defined(__has_builtin)
#if __has_builtin(__builtin_amdgcn_tensor_load_to_lds) && \
    __has_builtin(__builtin_amdgcn_s_wait_tensorcnt)
#define HAVE_TDM 1
#endif
#endif
#ifndef HAVE_TDM
#define HAVE_TDM 0
#endif

union AFrag {                 // 16x32 bf16 A/B fragment (8 VGPRs)
    v16bf v;
    unsigned short u[16];
    unsigned int   d[8];
};
union CFrag {                 // 16x16 f32 C/D fragment (8 VGPRs)
    v8f   v;
    float f[8];
};

__device__ __forceinline__ unsigned short f2bf(float f) {
    unsigned int u = __float_as_uint(f);
    u += 0x7FFFu + ((u >> 16) & 1u);     // round-to-nearest-even
    return (unsigned short)(u >> 16);
}

// K-offset of element e (0..15) of a 16-bit 16x32 fragment for lane-half hl.
__device__ __forceinline__ int kpack(int e, int hl) {
    int j = e >> 1;
    return ((j >> 2) << 4) + ((j & 3) << 1) + (e & 1) + (hl << 3);
}

// 16x32 bf16 fragment from row-major GLOBAL memory (K contiguous).
__device__ __forceinline__ void load_frag(AFrag& f, const unsigned short* __restrict__ base,
                                          int row0, int ld, int k0, int lane) {
    const int r  = row0 + (lane & 15);
    const int hl = lane >> 4;
    const unsigned short* p = base + (size_t)r * ld + k0 + (hl << 3);
#pragma unroll
    for (int j = 0; j < 8; ++j) {
        const int off = ((j >> 2) << 4) + ((j & 3) << 1);
        f.d[j] = *reinterpret_cast<const unsigned int*>(p + off);
    }
}

// 16x32 bf16 fragment from LDS tile (row pitch GEMM_PAD) -> two b128 loads.
__device__ __forceinline__ void load_frag_lds(AFrag& f, const unsigned short* rowbase, int lane) {
    const int r  = lane & 15;
    const int hl = lane >> 4;
    const unsigned short* p = rowbase + r * GEMM_PAD + (hl << 3);
    const uint4 lo = *reinterpret_cast<const uint4*>(p);        // elems 0..7
    const uint4 hi = *reinterpret_cast<const uint4*>(p + 16);   // elems 16..23
    f.d[0] = lo.x; f.d[1] = lo.y; f.d[2] = lo.z; f.d[3] = lo.w;
    f.d[4] = hi.x; f.d[5] = hi.y; f.d[6] = hi.z; f.d[7] = hi.w;
}

__device__ __forceinline__ v8f wmma_bf16(const AFrag& a, const AFrag& b, const CFrag& c) {
    return __builtin_amdgcn_wmma_f32_16x16x32_bf16(
        false, a.v, false, b.v, (short)0, c.v, false, false);
}

__device__ __forceinline__ unsigned int lds_off(const void* p) {
    return (unsigned int)(size_t)p;      // low 32 bits of generic LDS VA = LDS offset
}

#if HAVE_TDM
// TDM: DMA a 128-row x 32-elem bf16 tile (row stride ldg elems) from global
// into LDS at ldsByte, padding each 16-dword row with 4 dwords -> pitch 40.
__device__ __forceinline__ void tdm_load_tile(const unsigned short* g, int ldg,
                                              unsigned int ldsByte) {
    const unsigned long long ga = (unsigned long long)(size_t)g;
    u32x4 g0;
    g0[0] = 1u;                                              // count=1 (valid D#)
    g0[1] = ldsByte;                                         // lds_addr
    g0[2] = (unsigned int)ga;                                // global_addr[31:0]
    g0[3] = ((unsigned int)(ga >> 32) & 0x01FFFFFFu) | (2u << 30);  // ga[56:32] | type=2
    const unsigned int td0 = 1u << 20, td1 = 1u << 20;       // generous OOB dims
    const unsigned long long s0 = (unsigned long long)ldg;   // dim0 stride (elems)
    i32x8 g1;
    g1[0] = (1 << 16)    // data_size = 2B
          | (1 << 20)    // pad_enable
          | (3 << 22)    // pad_interval: every 16 dwords (one 32-elem row)
          | (3 << 25);   // pad_amount: 4 dwords (8 elems) -> pitch 40
    g1[1] = (int)((td0 & 0xFFFFu) << 16);                    // tensor_dim0 lo
    g1[2] = (int)((td0 >> 16) | ((td1 & 0xFFFFu) << 16));    // dim0 hi | dim1 lo
    g1[3] = (int)((td1 >> 16) | (32u << 16));                // dim1 hi | tile_dim0=32
    g1[4] = 128;                                             // tile_dim1=128, tile_dim2=0
    g1[5] = (int)(s0 & 0xFFFFFFFFu);
    g1[6] = (int)((s0 >> 32) & 0xFFFFu);                     // stride0 hi | stride1 lo
    g1[7] = 0;
    i32x4 gz4; gz4[0] = 0; gz4[1] = 0; gz4[2] = 0; gz4[3] = 0;
    i32x8 gz8;
#pragma unroll
    for (int i = 0; i < 8; ++i) gz8[i] = 0;
    // 6-arg form (clang-23 / therock headers): (g0, g1, g2, g3, g4, cpol)
    __builtin_amdgcn_tensor_load_to_lds(g0, g1, gz4, gz4, gz8, 0);
}
#else
// Fallback staging: 256 threads copy one 128x32 tile (8 KB) global->LDS.
__device__ __forceinline__ void stage_tile(unsigned short* dst, const unsigned short* src,
                                           int row0, int ldg, int k0) {
    const int t = threadIdx.x;
#pragma unroll
    for (int i = 0; i < 2; ++i) {
        const int idx  = t + i * 256;       // 0..511: 128 rows x 4 x 16B chunks
        const int row  = idx >> 2;
        const int ch   = idx & 3;
        const uint4 v = *reinterpret_cast<const uint4*>(
            src + (size_t)(row0 + row) * ldg + k0 + ch * 8);
        *reinterpret_cast<uint4*>(dst + row * GEMM_PAD + ch * 8) = v;
    }
}
#endif

// ---------------------------------------------------------------------------
__global__ __launch_bounds__(256)
void cvt_bf16(const float* __restrict__ in, unsigned short* __restrict__ out, int n) {
    int i = blockIdx.x * blockDim.x + threadIdx.x;
    if (i < n) out[i] = f2bf(in[i]);
}

// ---------------------------------------------------------------------------
// C[M,N] = A[M,K] * B[N,K]^T (row-major bf16, K contiguous), block tile
// 128x128, TDM double-buffered LDS staging, wave tile 64x32 (4x2 WMMA).
// outF -> fp32 (+bias); else bf16; storeT -> write C^T (outB[n*M+m]).
__global__ __launch_bounds__(256)
void gemm_nt_bf16(const unsigned short* __restrict__ A,
                  const unsigned short* __restrict__ B,
                  int M, int N, int K,
                  unsigned short* __restrict__ outB,
                  float* __restrict__ outF,
                  const float* __restrict__ bias,
                  int storeT) {
    __shared__ __align__(16) unsigned short ldsA[2][128 * GEMM_PAD];
    __shared__ __align__(16) unsigned short ldsB[2][128 * GEMM_PAD];

    const int lane = threadIdx.x & 31;
    const int wave = threadIdx.x >> 5;
    const int hl   = lane >> 4;
    const int col  = lane & 15;
    const int mBlk = blockIdx.y * 128;
    const int nBlk = blockIdx.x * 128;
    const int mW   = (wave & 1) * 64;     // wave tile: 64 rows x 32 cols
    const int nW   = (wave >> 1) * 32;

    CFrag acc[4][2];
#pragma unroll
    for (int i = 0; i < 4; ++i)
#pragma unroll
        for (int j = 0; j < 2; ++j)
#pragma unroll
            for (int v = 0; v < 8; ++v) acc[i][j].f[v] = 0.0f;

    const int nk = K / GEMM_KC;
#if HAVE_TDM
    if (wave == 0) tdm_load_tile(A + (size_t)mBlk * K, K, lds_off(ldsA[0]));
    if (wave == 1) tdm_load_tile(B + (size_t)nBlk * K, K, lds_off(ldsB[0]));
#endif

    for (int kt = 0; kt < nk; ++kt) {
        const unsigned short* sA;
        const unsigned short* sB;
#if HAVE_TDM
        const int buf = kt & 1;
        if (wave < 2) __builtin_amdgcn_s_wait_tensorcnt(0);
        __syncthreads();                        // tile(buf) landed; buf^1 reads done
        if (kt + 1 < nk) {
            const int k0 = (kt + 1) * GEMM_KC;
            if (wave == 0) tdm_load_tile(A + (size_t)mBlk * K + k0, K, lds_off(ldsA[buf ^ 1]));
            if (wave == 1) tdm_load_tile(B + (size_t)nBlk * K + k0, K, lds_off(ldsB[buf ^ 1]));
        }
        sA = ldsA[buf]; sB = ldsB[buf];
#else
        __syncthreads();                        // previous reads done
        stage_tile(ldsA[0], A, mBlk, K, kt * GEMM_KC);
        stage_tile(ldsB[0], B, nBlk, K, kt * GEMM_KC);
        __syncthreads();
        sA = ldsA[0]; sB = ldsB[0];
#endif
        AFrag a[4], b[2];
#pragma unroll
        for (int ti = 0; ti < 4; ++ti)
            load_frag_lds(a[ti], sA + (mW + ti * 16) * GEMM_PAD, lane);
#pragma unroll
        for (int tj = 0; tj < 2; ++tj)
            load_frag_lds(b[tj], sB + (nW + tj * 16) * GEMM_PAD, lane);
#pragma unroll
        for (int ti = 0; ti < 4; ++ti)
#pragma unroll
            for (int tj = 0; tj < 2; ++tj)
                acc[ti][tj].v = wmma_bf16(a[ti], b[tj], acc[ti][tj]);
    }

#pragma unroll
    for (int ti = 0; ti < 4; ++ti)
#pragma unroll
        for (int tj = 0; tj < 2; ++tj)
#pragma unroll
            for (int v = 0; v < 8; ++v) {
                const int m = mBlk + mW + ti * 16 + (hl << 3) + v;
                const int n = nBlk + nW + tj * 16 + col;
                const float r = acc[ti][tj].f[v];
                if (outF)        outF[(size_t)m * N + n] = r + (bias ? bias[n] : 0.0f);
                else if (storeT) outB[(size_t)n * M + m] = f2bf(r);
                else             outB[(size_t)m * N + n] = f2bf(r);
            }
}

// ---------------------------------------------------------------------------
// Flash-style causal GQA attention. One wave owns (head h, 16 query rows).
// Q,K bf16 row-major; V pre-transposed (Vt[d, t], t contiguous); ctx bf16.
__global__ __launch_bounds__(128)
void gqa_attention(const unsigned short* __restrict__ Q,
                   const unsigned short* __restrict__ Kb,
                   const unsigned short* __restrict__ Vt,
                   unsigned short* __restrict__ Cb) {
    __shared__ unsigned short plds[4][16][36];   // wave-private P staging (padded)

    const int lane = threadIdx.x & 31;
    const int wave = threadIdx.x >> 5;
    const int hl   = lane >> 4;
    const int col  = lane & 15;
    const int qb   = blockIdx.x * 4 + wave;
    const int h    = blockIdx.y;
    const int kvB  = (h >> 2) * HDIM;            // shared-KV head base channel
    const int m0   = qb * 16;

    AFrag qa[2];
    load_frag(qa[0], Q, m0, MDIM, h * HDIM,      lane);
    load_frag(qa[1], Q, m0, MDIM, h * HDIM + 32, lane);

    float row_max[8], row_sum[8];
    CFrag o[4];
#pragma unroll
    for (int v = 0; v < 8; ++v) { row_max[v] = -3.0e38f; row_sum[v] = 0.0f; }
#pragma unroll
    for (int nt = 0; nt < 4; ++nt)
#pragma unroll
        for (int v = 0; v < 8; ++v) o[nt].f[v] = 0.0f;

    for (int t0 = 0; t0 < m0 + 16; t0 += 32) {   // 32-key chunks, causal bound
        CFrag s[2];
#pragma unroll
        for (int sub = 0; sub < 2; ++sub) {
            const int t16 = t0 + sub * 16;
            if (t16 < m0 + 16) {
#pragma unroll
                for (int v = 0; v < 8; ++v) s[sub].f[v] = 0.0f;
                AFrag kf;
                load_frag(kf, Kb, t16, KVDIM, kvB,      lane);
                s[sub].v = wmma_bf16(qa[0], kf, s[sub]);
                load_frag(kf, Kb, t16, KVDIM, kvB + 32, lane);
                s[sub].v = wmma_bf16(qa[1], kf, s[sub]);
#pragma unroll
                for (int v = 0; v < 8; ++v) {    // scale 1/sqrt(64) + causal mask
                    const int m = m0 + (hl << 3) + v;
                    const int t = t16 + col;
                    const float val = s[sub].f[v] * 0.125f;
                    s[sub].f[v] = (t <= m) ? val : -1.0e9f;
                }
            } else {
#pragma unroll
                for (int v = 0; v < 8; ++v) s[sub].f[v] = -1.0e9f;
            }
        }

        // online softmax update (v indexes row m0 + hl*8 + v)
#pragma unroll
        for (int v = 0; v < 8; ++v) {
            float mx = fmaxf(s[0].f[v], s[1].f[v]);
#pragma unroll
            for (int d = 1; d < 16; d <<= 1) mx = fmaxf(mx, __shfl_xor(mx, d, 32));
            const float nm = fmaxf(row_max[v], mx);
            const float sc = __expf(row_max[v] - nm);
            row_max[v] = nm;
            const float p0 = __expf(s[0].f[v] - nm);
            const float p1 = __expf(s[1].f[v] - nm);
            s[0].f[v] = p0; s[1].f[v] = p1;
            float ps = p0 + p1;
#pragma unroll
            for (int d = 1; d < 16; d <<= 1) ps += __shfl_xor(ps, d, 32);
            row_sum[v] = row_sum[v] * sc + ps;
#pragma unroll
            for (int nt = 0; nt < 4; ++nt) o[nt].f[v] *= sc;
        }

        // stage P (bf16) through LDS: C-fragment layout -> A-fragment layout
#pragma unroll
        for (int sub = 0; sub < 2; ++sub)
#pragma unroll
            for (int v = 0; v < 8; ++v)
                plds[wave][(hl << 3) + v][sub * 16 + col] = f2bf(s[sub].f[v]);
        asm volatile("s_wait_dscnt 0" ::: "memory");

        AFrag pa;
#pragma unroll
        for (int e = 0; e < 16; ++e)
            pa.u[e] = plds[wave][lane & 15][kpack(e, hl)];

        // o[nt] += P(16x32 keys) x V(32 keys x 16 dims); Vt rows = channels
#pragma unroll
        for (int nt = 0; nt < 4; ++nt) {
            AFrag vf;
            load_frag(vf, Vt, kvB + nt * 16, S_LEN, t0, lane);
            o[nt].v = wmma_bf16(pa, vf, o[nt]);
        }
    }

#pragma unroll
    for (int v = 0; v < 8; ++v) {
        const float inv = 1.0f / row_sum[v];
        const int m = m0 + (hl << 3) + v;
#pragma unroll
        for (int nt = 0; nt < 4; ++nt)
            Cb[(size_t)m * MDIM + h * HDIM + nt * 16 + col] = f2bf(o[nt].f[v] * inv);
    }
}

// ---------------------------------------------------------------------------
extern "C" void kernel_launch(void* const* d_in, const int* in_sizes, int n_in,
                              void* d_out, int out_size, void* d_ws, size_t ws_size,
                              hipStream_t stream) {
    const float* x   = (const float*)d_in[0];
    // d_in[1] = causal mask (tril) -- computed analytically in-kernel
    const float* w_q = (const float*)d_in[2];
    const float* w_k = (const float*)d_in[3];
    const float* w_v = (const float*)d_in[4];
    const float* w_o = (const float*)d_in[5];
    const float* b_o = (const float*)d_in[6];
    float* out = (float*)d_out;

    char* ws = (char*)d_ws;
    size_t off = 0;
    auto carve = [&](size_t elems) {
        void* p = ws + off;
        off = (off + elems * sizeof(unsigned short) + 255) & ~(size_t)255;
        return (unsigned short*)p;
    };
    unsigned short* xb  = carve((size_t)S_LEN * MDIM);
    unsigned short* wqb = carve((size_t)MDIM * MDIM);
    unsigned short* wkb = carve((size_t)KVDIM * MDIM);
    unsigned short* wvb = carve((size_t)KVDIM * MDIM);
    unsigned short* wob = carve((size_t)MDIM * MDIM);
    unsigned short* Qb  = carve((size_t)S_LEN * MDIM);
    unsigned short* Kb  = carve((size_t)S_LEN * KVDIM);
    unsigned short* Vtb = carve((size_t)KVDIM * S_LEN);   // transposed V [d, t]
    unsigned short* Cb  = carve((size_t)S_LEN * MDIM);

    const int CV = 256;
    auto cvtN = [&](const float* src, unsigned short* dst, int n) {
        cvt_bf16<<<(n + CV - 1) / CV, CV, 0, stream>>>(src, dst, n);
    };
    cvtN(x,   xb,  S_LEN * MDIM);
    cvtN(w_q, wqb, MDIM * MDIM);
    cvtN(w_k, wkb, KVDIM * MDIM);
    cvtN(w_v, wvb, KVDIM * MDIM);
    cvtN(w_o, wob, MDIM * MDIM);

    // Q = x @ w_q^T  [2048 x 2048]
    gemm_nt_bf16<<<dim3(MDIM / 128, S_LEN / 128), 256, 0, stream>>>(
        xb, wqb, S_LEN, MDIM, MDIM, Qb, nullptr, nullptr, 0);
    // K = x @ w_k^T  [2048 x 512]
    gemm_nt_bf16<<<dim3(KVDIM / 128, S_LEN / 128), 256, 0, stream>>>(
        xb, wkb, S_LEN, KVDIM, MDIM, Kb, nullptr, nullptr, 0);
    // V^T = (x @ w_v^T)^T  [512 x 2048] (t contiguous for attention PV)
    gemm_nt_bf16<<<dim3(KVDIM / 128, S_LEN / 128), 256, 0, stream>>>(
        xb, wvb, S_LEN, KVDIM, MDIM, Vtb, nullptr, nullptr, 1);

    // attention: grid (query blocks / 4 waves, heads), 128 thr = 4 waves
    gqa_attention<<<dim3(S_LEN / 16 / 4, NHEADS), 128, 0, stream>>>(Qb, Kb, Vtb, Cb);

    // out = ctx @ w_o^T + b_o  (fp32)
    gemm_nt_bf16<<<dim3(MDIM / 128, S_LEN / 128), 256, 0, stream>>>(
        Cb, wob, S_LEN, MDIM, MDIM, nullptr, out, b_o, 0);
}
